// Neural_Expert_SDE_integrator_86157043958009
// MI455X (gfx1250) — compile-verified, matrix operand order checked by reading, not run on hardware
//
#include <hip/hip_runtime.h>
#include <hip/hip_bf16.h>

typedef __bf16 v16bf __attribute__((ext_vector_type(16)));
typedef float  v8f   __attribute__((ext_vector_type(8)));

#define NROW   16384
#define DIMA   22
#define LAT    20
#define HID    80
#define KP     96            // padded K pitch in halfs (3 chunks of 32)
#define DTV    0.05f
#define SGSQ   0.1118033989f // SIGMA * sqrt(dt) = 0.5*sqrt(0.05)
#define NFRAG  71            // 5 (in) + 4*15 (hidden) + 6 (out)
#define WPACK_OFF 1024       // byte offset of packed weights in d_ws
#define YS_ELEMS ((size_t)256*64*41*20)

union FragU { uint4 u[2]; v16bf v; };

__device__ __forceinline__ unsigned short f2bf_bits(float x){
  unsigned int u = __float_as_uint(x);
  u += 0x7FFFu + ((u >> 16) & 1u);
  return (unsigned short)(u >> 16);
}

// A fragment (16x32 bf16, MxK) from LDS activation panel, per ISA 7.12.2 layout.
__device__ __forceinline__ v16bf loadA(const __bf16* actw, int lane, int kc){
  int m  = lane & 15;
  int k0 = kc*32 + ((lane >> 4) << 3);
  FragU f;
  f.u[0] = *(const uint4*)(actw + m*KP + k0);
  f.u[1] = *(const uint4*)(actw + m*KP + k0 + 16);
  return f.v;
}

// B fragment (32x16 bf16, KxN) from fragment-linear packed weights in global.
__device__ __forceinline__ v16bf loadB(const unsigned short* wp, int frag, int lane){
  const uint4* p = (const uint4*)(wp + frag*512 + lane*16);
  FragU f; f.u[0] = p[0]; f.u[1] = p[1]; return f.v;
}

__device__ __forceinline__ v8f bwmma(v16bf a, v16bf b, v8f c){
  return __builtin_amdgcn_wmma_f32_16x16x32_bf16(false, a, false, b, (short)0, c, false, false);
}

// ---------------- time-MLP precompute: u(t_i) for i = 0..39 -----------------
__global__ void uprep_kernel(const float* __restrict__ t,
                             const float* __restrict__ twi, const float* __restrict__ tbi,
                             const float* __restrict__ twh, const float* __restrict__ tbh,
                             const float* __restrict__ two, const float* __restrict__ tbo,
                             float* __restrict__ uvals){
  int i = blockIdx.x*blockDim.x + threadIdx.x;
  if (i >= 40) return;
  float ti = t[i];
  float h[20], h2[20];
  for (int j=0;j<20;++j) h[j] = fmaxf(twi[j]*ti + tbi[j], 0.f);
  for (int l=0;l<4;++l){
    for (int j=0;j<20;++j){
      float s = tbh[l*20+j];
      for (int k=0;k<20;++k) s += twh[l*400 + j*20 + k]*h[k];
      h2[j] = fmaxf(s, 0.f);
    }
    for (int j=0;j<20;++j) h[j] = h2[j];
  }
  for (int o=0;o<4;++o){
    float s = tbo[o];
    for (int k=0;k<20;++k) s += two[o*20+k]*h[k];
    uvals[i*4+o] = s;
  }
}

// -------- pack drift-MLP weights into WMMA B-fragment-linear bf16 layout ----
__global__ void pack_kernel(const float* __restrict__ dwin,
                            const float* __restrict__ dwh,
                            const float* __restrict__ dwo,
                            unsigned short* __restrict__ wp){
  int g = blockIdx.x*blockDim.x + threadIdx.x;
  if (g >= NFRAG*512) return;
  int frag = g >> 9;
  int w    = g & 511;
  int lane = w >> 4;
  int j    = w & 15;
  int layer, tile, kc;
  if (frag < 5)      { layer = 0; tile = frag; kc = 0; }
  else if (frag < 65){ int f = frag-5; layer = 1 + f/15; f %= 15; tile = f/3; kc = f%3; }
  else               { int f = frag-65; layer = 5; tile = f/3; kc = f%3; }
  int n = tile*16 + (lane & 15);
  int k = kc*32 + (j & 7) + ((j >> 3) << 4) + ((lane >> 4) << 3);
  float v = 0.f;
  if (layer == 0)     { if (k < 24)             v = dwin[n*24 + k]; }
  else if (layer < 5) { if (k < HID)            v = dwh[(layer-1)*6400 + n*HID + k]; }
  else                { if (k < HID && n < LAT) v = dwo[n*HID + k]; }
  wp[g] = f2bf_bits(v);
}

// ------------------------- persistent SDE integrator ------------------------
__global__ void __launch_bounds__(256, 1)
sde_kernel(const float* __restrict__ z,    const float* __restrict__ Tx,
           const float* __restrict__ noise,
           const float* __restrict__ dbin, const float* __restrict__ dbh,
           const float* __restrict__ dbo,
           const float* __restrict__ uvals, const unsigned short* __restrict__ wp,
           float* __restrict__ ys, float* __restrict__ lq){
  __shared__ float ystate[128][DIMA];                 // 11.0 KB
  __shared__ __align__(16) __bf16 act[8][16*KP];      // 24.0 KB
  __shared__ float fbuf[8][16*LAT];                   // 10.0 KB
  __shared__ float uub [8][16*LAT];                   // 10.0 KB

  const int w     = threadIdx.x >> 5;
  const int lane  = threadIdx.x & 31;
  const int rbase = blockIdx.x*128 + w*16;  // global row index of this wave's m=0
  const int w16   = w*16;
  __bf16* actw = act[w];
  float* fb = fbuf[w];
  float* uu = uub[w];

  // ---- init state (each wave owns its 16 rows; no cross-wave sharing) ----
  for (int e = lane; e < 16*DIMA; e += 32){
    int m = e/DIMA, d = e%DIMA, n = rbase + m;
    float v;
    if (d < LAT)       v = z[n*LAT + d];
    else if (d == LAT) v = 0.f;
    else               v = Tx[n >> 6];
    ystate[w16+m][d] = v;
  }
  // zero the K-pad columns 80..95 once (never overwritten later)
  for (int e = lane; e < 16*16; e += 32){
    int m = e >> 4, k = HID + (e & 15);
    actw[m*KP + k] = (__bf16)0.f;
  }
  // t = 0 outputs
  for (int e = lane; e < 16*LAT; e += 32){
    int m = e/LAT, d = e%LAT, n = rbase + m;
    ys[(size_t)n*LAT + d] = z[n*LAT + d];
  }
  for (int m = lane; m < 16; m += 32){
    int n = rbase + m;
    if (n % 41 == 40) lq[n / 41] = 0.f;
  }

  for (int st = 0; st < 40; ++st){
    // prefetch this step's noise rows (global_prefetch_b8)
    __builtin_prefetch(noise + (size_t)st*NROW*DIMA + (size_t)(rbase + (lane & 15))*DIMA, 0, 1);

    float u0 = uvals[st*4+0], u1 = uvals[st*4+1];
    float u2 = uvals[st*4+2], u3 = uvals[st*4+3];

    // ---- build input activations: [y(0..19) | u*T (20..23) | 0 (24..31)] ----
    for (int e = lane; e < 16*32; e += 32){
      int m = e >> 5, k = e & 31;
      float v;
      if (k < LAT) v = ystate[w16+m][k];
      else if (k < 24){
        float T = ystate[w16+m][21];
        v = (k == 20 ? u0 : k == 21 ? u1 : k == 22 ? u2 : u3) * T;
      } else v = 0.f;
      actw[m*KP + k] = (__bf16)v;
    }

    // ---- input layer: 24(->32) -> 80, ReLU ----
    {
      v16bf A0 = loadA(actw, lane, 0);
      v8f acc[5];
      #pragma unroll
      for (int t = 0; t < 5; ++t){
        v8f c = {0.f,0.f,0.f,0.f,0.f,0.f,0.f,0.f};
        acc[t] = bwmma(A0, loadB(wp, t, lane), c);
      }
      #pragma unroll
      for (int t = 0; t < 5; ++t){
        float b = dbin[t*16 + (lane & 15)];
        #pragma unroll
        for (int r = 0; r < 8; ++r){
          float x = fmaxf(acc[t][r] + b, 0.f);
          actw[(r + ((lane >> 4) << 3))*KP + t*16 + (lane & 15)] = (__bf16)x;
        }
      }
    }

    // ---- 4 hidden layers: 80 -> 80, tanh ----
    for (int L = 0; L < 4; ++L){
      v16bf A0 = loadA(actw, lane, 0);
      v16bf A1 = loadA(actw, lane, 1);
      v16bf A2 = loadA(actw, lane, 2);
      int f0 = 5 + L*15;
      v8f acc[5];
      #pragma unroll
      for (int t = 0; t < 5; ++t){
        v8f c = {0.f,0.f,0.f,0.f,0.f,0.f,0.f,0.f};
        c = bwmma(A0, loadB(wp, f0 + t*3 + 0, lane), c);
        c = bwmma(A1, loadB(wp, f0 + t*3 + 1, lane), c);
        c = bwmma(A2, loadB(wp, f0 + t*3 + 2, lane), c);
        acc[t] = c;
      }
      #pragma unroll
      for (int t = 0; t < 5; ++t){
        float b = dbh[L*HID + t*16 + (lane & 15)];
        #pragma unroll
        for (int r = 0; r < 8; ++r){
          float x = tanhf(acc[t][r] + b);
          actw[(r + ((lane >> 4) << 3))*KP + t*16 + (lane & 15)] = (__bf16)x;
        }
      }
    }

    // ---- output layer: 80 -> 20 (no activation) ----
    {
      v16bf A0 = loadA(actw, lane, 0);
      v16bf A1 = loadA(actw, lane, 1);
      v16bf A2 = loadA(actw, lane, 2);
      #pragma unroll
      for (int t = 0; t < 2; ++t){
        v8f c = {0.f,0.f,0.f,0.f,0.f,0.f,0.f,0.f};
        c = bwmma(A0, loadB(wp, 65 + t*3 + 0, lane), c);
        c = bwmma(A1, loadB(wp, 65 + t*3 + 1, lane), c);
        c = bwmma(A2, loadB(wp, 65 + t*3 + 2, lane), c);
        int n = t*16 + (lane & 15);
        if (n < LAT){
          float b = dbo[n];
          #pragma unroll
          for (int r = 0; r < 8; ++r)
            fb[(r + ((lane >> 4) << 3))*LAT + n] = c[r] + b;
        }
      }
    }

    // ---- Euler-Maruyama update + output write ----
    // theta=1, mu=0: h=-y, f=p+y, uu=(f-h)/sigma=2(p+2y), f_logqp=2*sum((p+2y)^2)
    int tcur = st + 1;
    for (int e = lane; e < 16*LAT; e += 32){
      int m = e/LAT, d = e%LAT;
      int row = w16 + m, n = rbase + m;
      float y  = ystate[row][d];
      float p  = fb[m*LAT + d];
      float f  = p + y;
      float dW = noise[(size_t)st*NROW*DIMA + (size_t)n*DIMA + d];
      float yn = y + f*DTV + SGSQ*dW;
      ystate[row][d] = yn;
      float s = p + 2.f*y;
      uu[m*LAT + d] = s*s;
      ys[(size_t)(tcur*NROW + n)*LAT + d] = yn;
    }
    if (lane < 16){
      int m = lane, row = w16 + m, n = rbase + m;
      float ssum = 0.f;
      for (int d = 0; d < LAT; ++d) ssum += uu[m*LAT + d];
      float l = ystate[row][LAT] + 2.f*ssum*DTV;
      ystate[row][LAT] = l;
      int G = tcur*NROW + n;
      if (G % 41 == 40) lq[G / 41] = l;
    }
  }
}

extern "C" void kernel_launch(void* const* d_in, const int* in_sizes, int n_in,
                              void* d_out, int out_size, void* d_ws, size_t ws_size,
                              hipStream_t stream){
  const float* z     = (const float*)d_in[0];
  const float* t     = (const float*)d_in[1];
  const float* Tx    = (const float*)d_in[2];
  const float* noise = (const float*)d_in[3];
  const float* twi   = (const float*)d_in[4];
  const float* tbi   = (const float*)d_in[5];
  const float* twh   = (const float*)d_in[6];
  const float* tbh   = (const float*)d_in[7];
  const float* two   = (const float*)d_in[8];
  const float* tbo   = (const float*)d_in[9];
  const float* dwin  = (const float*)d_in[10];
  const float* dbin  = (const float*)d_in[11];
  const float* dwh   = (const float*)d_in[12];
  const float* dbh   = (const float*)d_in[13];
  const float* dwo   = (const float*)d_in[14];
  const float* dbo   = (const float*)d_in[15];

  float* uvals = (float*)d_ws;
  unsigned short* wpack = (unsigned short*)((char*)d_ws + WPACK_OFF);
  float* ys = (float*)d_out;
  float* lq = ys + YS_ELEMS;

  uprep_kernel<<<1, 64, 0, stream>>>(t, twi, tbi, twh, tbh, two, tbo, uvals);
  pack_kernel<<<(NFRAG*512 + 255)/256, 256, 0, stream>>>(dwin, dwh, dwo, wpack);
  sde_kernel<<<128, 256, 0, stream>>>(z, Tx, noise, dbin, dbh, dbo, uvals, wpack, ys, lq);
}
